// MMSingleStreamBlock_6502580486393
// MI455X (gfx1250) — matrix operand order repaired and checked by reading, not compiled, not run on hardware
//
#include <hip/hip_runtime.h>
#include <hip/hip_bf16.h>

// ---------------- problem constants ----------------
#define S_LEN   2048
#define HIDDEN  3072
#define N_HEADS 24
#define HEAD_D  128
#define MLP_D   12288
#define QKV_D   9216        // 3*HIDDEN
#define H1_D    21504       // 3*HIDDEN + MLP_DIM
#define COMB_D  15360       // HIDDEN + MLP_DIM
#define IMG_LEN 1792
#define EPSV    1e-6f

// ---------------- vector types ----------------
typedef __attribute__((ext_vector_type(16))) __bf16 bf16x16;
typedef __attribute__((ext_vector_type(8)))  __bf16 bf16x8;
typedef __attribute__((ext_vector_type(8)))  float  f32x8;
typedef __attribute__((ext_vector_type(4)))  float  f32x4;

union BF16x16 { bf16x16 v; bf16x8 h[2]; };

__device__ __forceinline__ f32x8 zero8() {
  f32x8 z = {0.f,0.f,0.f,0.f,0.f,0.f,0.f,0.f};
  return z;
}

__device__ __forceinline__ f32x8 wmma_bf16(bf16x16 a, bf16x16 b, f32x8 c) {
  // D = A(16x32 bf16) x B(32x16 bf16) + C(16x16 f32)
  return __builtin_amdgcn_wmma_f32_16x16x32_bf16(
      /*neg_a=*/false, a, /*neg_b=*/false, b,
      /*c_mod=*/(short)0, c, /*reuse_a=*/false, /*reuse_b=*/false);
}

// pack two floats -> one dword of two bf16 (lowers to v_cvt_pk_bf16_f32)
__device__ __forceinline__ unsigned pack2bf(float lo, float hi) {
  union { __bf16 b[2]; unsigned u; } p;
  p.b[0] = (__bf16)lo;
  p.b[1] = (__bf16)hi;
  return p.u;
}

// =====================================================================
// K1: mod = silu(vec) @ w_mod.T + b_mod    (9216 outputs, GEMV)
// =====================================================================
__global__ void mod_gemv_kernel(const float* __restrict__ vec,
                                const float* __restrict__ w_mod,
                                const float* __restrict__ b_mod,
                                float* __restrict__ mod) {
  const int row  = blockIdx.x * 8 + (threadIdx.x >> 5);
  const int lane = threadIdx.x & 31;
  const float* wr = w_mod + (size_t)row * HIDDEN;
  float sum = 0.f;
  for (int k = lane * 4; k < HIDDEN; k += 32 * 4) {
    f32x4 w4 = *(const f32x4*)(wr + k);
    f32x4 v4 = *(const f32x4*)(vec + k);
    float s0 = v4.x / (1.f + __expf(-v4.x));
    float s1 = v4.y / (1.f + __expf(-v4.y));
    float s2 = v4.z / (1.f + __expf(-v4.z));
    float s3 = v4.w / (1.f + __expf(-v4.w));
    sum += w4.x * s0 + w4.y * s1 + w4.z * s2 + w4.w * s3;
  }
#pragma unroll
  for (int o = 16; o > 0; o >>= 1) sum += __shfl_xor(sum, o, 32);
  if (lane == 0) mod[row] = sum + b_mod[row];
}

// =====================================================================
// K2: layernorm + modulation, fp32 -> bf16 A-matrix  (2048 x 3072)
// =====================================================================
__global__ void ln_mod_kernel(const float* __restrict__ x,
                              const float* __restrict__ mod,
                              __bf16* __restrict__ xmod) {
  const int row = blockIdx.x;
  const float* xr = x + (size_t)row * HIDDEN;
  __shared__ float red[8];
  const int tid = threadIdx.x, wv = tid >> 5, ln = tid & 31;

  float s = 0.f;
  for (int k = tid * 4; k < HIDDEN; k += 1024) {
    f32x4 v = *(const f32x4*)(xr + k);
    s += v.x + v.y + v.z + v.w;
  }
#pragma unroll
  for (int o = 16; o > 0; o >>= 1) s += __shfl_xor(s, o, 32);
  if (ln == 0) red[wv] = s;
  __syncthreads();
  s = red[0]+red[1]+red[2]+red[3]+red[4]+red[5]+red[6]+red[7];
  const float mu = s * (1.f / HIDDEN);
  __syncthreads();

  float vs = 0.f;
  for (int k = tid * 4; k < HIDDEN; k += 1024) {
    f32x4 v = *(const f32x4*)(xr + k);
    float a = v.x - mu, b = v.y - mu, c = v.z - mu, d = v.w - mu;
    vs += a*a + b*b + c*c + d*d;
  }
#pragma unroll
  for (int o = 16; o > 0; o >>= 1) vs += __shfl_xor(vs, o, 32);
  if (ln == 0) red[wv] = vs;
  __syncthreads();
  vs = red[0]+red[1]+red[2]+red[3]+red[4]+red[5]+red[6]+red[7];
  const float rstd = rsqrtf(vs * (1.f / HIDDEN) + EPSV);

  for (int k = tid * 4; k < HIDDEN; k += 1024) {
    f32x4 v = *(const f32x4*)(xr + k);
#pragma unroll
    for (int j = 0; j < 4; j++) {
      float lnv = (v[j] - mu) * rstd;
      float o = lnv * (1.f + mod[HIDDEN + k + j]) + mod[k + j];
      xmod[(size_t)row * HIDDEN + k + j] = (__bf16)o;
    }
  }
}

// =====================================================================
// GEMM core: C(BMxBN) = A_bf16(MxK, K-major) x W_fp32(NxK, K-major)^T
// Tiles: BM=128 BN=128 BK=32; 8 waves (4 M x 2 N); wave = 2x4 subtiles.
// A read straight from global (L2-resident); W staged fp32->bf16 into a
// double-buffered LDS tile in WMMA B layout (lane<->N, VGPR<->K), with
// paired-column packing so every LDS store is a full dword, plus a
// 2-tiles-ahead global prefetch on the HBM-streamed weight operand.
// =====================================================================
#define BM 128
#define BN 128
#define BK 32
#define LDSB_P 136   // ushort pitch, 16B-aligned rows

__device__ __forceinline__ void gemm_core(const __bf16* __restrict__ A,
                                          const float*  __restrict__ W,
                                          int K, int blockM, int blockN,
                                          __bf16 (*ldsB)[BK][LDSB_P],
                                          f32x8 (&acc)[2][4]) {
  const int tid   = threadIdx.x;
  const int lane  = tid & 31;
  const int wave  = tid >> 5;
  const int waveM = wave & 3;
  const int waveN = wave >> 2;
#pragma unroll
  for (int i = 0; i < 2; i++)
#pragma unroll
    for (int j = 0; j < 4; j++) acc[i][j] = zero8();

  // weight staging assignment: each thread owns column pair (nPair, nPair+1)
  // and an 8-wide k chunk; packs two bf16 per LDS dword store.
  const int nPair = (tid & 63) * 2;
  const int kc    = (tid >> 6) * 8;
  const float* Wr0 = W + (size_t)(blockN * BN + nPair) * K;
  const float* Wr1 = Wr0 + K;

  const int aRow0 = blockM * BM + waveM * 32 + (lane & 15);
  const int kHalf = (lane >> 4) * 8;

  auto stage = [&](int buf, int k0) {
    f32x4 a0 = *(const f32x4*)(Wr0 + k0 + kc);
    f32x4 a1 = *(const f32x4*)(Wr0 + k0 + kc + 4);
    f32x4 b0 = *(const f32x4*)(Wr1 + k0 + kc);
    f32x4 b1 = *(const f32x4*)(Wr1 + k0 + kc + 4);
#pragma unroll
    for (int j = 0; j < 4; j++) {
      *(unsigned*)&ldsB[buf][kc + j][nPair]     = pack2bf(a0[j], b0[j]);
      *(unsigned*)&ldsB[buf][kc + 4 + j][nPair] = pack2bf(a1[j], b1[j]);
    }
  };

  int buf = 0;
  stage(0, 0);
  for (int k0 = 0; k0 < K; k0 += BK) {
    __syncthreads();                       // stores to `buf` visible, prior
                                           // readers of buf^1 are done
    if (k0 + BK < K) stage(buf ^ 1, k0 + BK);
    if (k0 + 2 * BK < K)
      __builtin_prefetch(Wr0 + k0 + 2 * BK, 0, 3);   // global_prefetch_b8

    // A fragments, direct from global (16-bit A layout)
    BF16x16 af[2];
#pragma unroll
    for (int mi = 0; mi < 2; mi++) {
      const __bf16* ap = A + (size_t)(aRow0 + mi * 16) * K + k0 + kHalf;
      af[mi].h[0] = *(const bf16x8*)(ap);
      af[mi].h[1] = *(const bf16x8*)(ap + 16);
    }
    // B fragments from LDS: lane = N column, 16 K values in table order
#pragma unroll
    for (int ni = 0; ni < 4; ni++) {
      const __bf16* bp = &ldsB[buf][kHalf][waveN * 64 + ni * 16 + (lane & 15)];
      BF16x16 bf;
#pragma unroll
      for (int j = 0; j < 8; j++) {
        bf.v[j]     = bp[j * LDSB_P];
        bf.v[8 + j] = bp[(16 + j) * LDSB_P];
      }
      acc[0][ni] = wmma_bf16(af[0].v, bf.v, acc[0][ni]);
      acc[1][ni] = wmma_bf16(af[1].v, bf.v, acc[1][ni]);
    }
    buf ^= 1;
  }
}

// =====================================================================
// K3: h1 = x_mod @ w1.T + b1 ; split -> qkv(fp32) | gelu(mlp)->bf16
// =====================================================================
__global__ void gemm1_kernel(const __bf16* __restrict__ A,
                             const float* __restrict__ w1,
                             const float* __restrict__ b1,
                             float* __restrict__ qkv,
                             __bf16* __restrict__ combined) {
  __shared__ __bf16 ldsB[2][BK][LDSB_P];
  f32x8 acc[2][4];
  gemm_core(A, w1, HIDDEN, blockIdx.y, blockIdx.x, ldsB, acc);

  const int lane = threadIdx.x & 31;
  const int wave = threadIdx.x >> 5;
  const int mBase = blockIdx.y * BM + (wave & 3) * 32 + 8 * (lane >> 4);
  const int nBase = blockIdx.x * BN + (wave >> 2) * 64 + (lane & 15);
#pragma unroll
  for (int mi = 0; mi < 2; mi++)
#pragma unroll
    for (int ni = 0; ni < 4; ni++) {
      const int n = nBase + ni * 16;
      const float bias = b1[n];
#pragma unroll
      for (int r = 0; r < 8; r++) {
        const int m = mBase + mi * 16 + r;
        float v = acc[mi][ni][r] + bias;
        if (n < QKV_D) {
          qkv[(size_t)m * QKV_D + n] = v;
        } else {
          float g = 0.5f * v * (1.f + tanhf(0.7978845608028654f *
                                            (v + 0.044715f * v * v * v)));
          combined[(size_t)m * COMB_D + HIDDEN + (n - QKV_D)] = (__bf16)g;
        }
      }
    }
}

// =====================================================================
// K4: per-(token,head) RMSNorm on q,k + interleaved RoPE (img tokens),
//     write qT (feature-major, pre-scaled), k and v (row-major) in bf16
// =====================================================================
__global__ void qk_rope_kernel(const float* __restrict__ qkv,
                               const float* __restrict__ qw,
                               const float* __restrict__ kw,
                               const float* __restrict__ cosT,
                               const float* __restrict__ sinT,
                               __bf16* __restrict__ qT,
                               __bf16* __restrict__ kB,
                               __bf16* __restrict__ vB) {
  const int s = blockIdx.x;      // token
  const int h = blockIdx.y;      // head
  const int d = threadIdx.x;     // feature 0..127
  const float* base = qkv + (size_t)s * QKV_D + h * HEAD_D + d;
  float q = base[0];
  float k = base[HIDDEN];
  float v = base[2 * HIDDEN];

  __shared__ float redq[4], redk[4];
  float sq = q * q, sk = k * k;
#pragma unroll
  for (int o = 16; o > 0; o >>= 1) {
    sq += __shfl_xor(sq, o, 32);
    sk += __shfl_xor(sk, o, 32);
  }
  const int wv = threadIdx.x >> 5, ln = threadIdx.x & 31;
  if (ln == 0) { redq[wv] = sq; redk[wv] = sk; }
  __syncthreads();
  sq = redq[0] + redq[1] + redq[2] + redq[3];
  sk = redk[0] + redk[1] + redk[2] + redk[3];
  q *= rsqrtf(sq * (1.f / HEAD_D) + EPSV) * qw[d];
  k *= rsqrtf(sk * (1.f / HEAD_D) + EPSV) * kw[d];

  if (s < IMG_LEN) {   // interleaved RoPE, partner lane = d^1
    const int i = d >> 1;
    const float c  = cosT[s * (HEAD_D / 2) + i];
    const float sn = sinT[s * (HEAD_D / 2) + i];
    const float qp = __shfl_xor(q, 1, 32);
    const float kp = __shfl_xor(k, 1, 32);
    if ((d & 1) == 0) { q = q * c - qp * sn; k = k * c - kp * sn; }
    else              { q = q * c + qp * sn; k = k * c + kp * sn; }
  }
  qT[((size_t)h * HEAD_D + d) * S_LEN + s] = (__bf16)(q * 0.08838834764831845f);
  kB[((size_t)h * S_LEN + s) * HEAD_D + d] = (__bf16)k;
  vB[((size_t)h * S_LEN + s) * HEAD_D + d] = (__bf16)v;
}

// =====================================================================
// K5: flash attention. Per block: one head, 128 query rows (wave = 16).
// S^T = K x q^T (C-layout lands directly in P's A-fragment layout),
// online softmax across 64 blocks of 32 keys, then P x V.
// =====================================================================
__global__ void attn_kernel(const __bf16* __restrict__ qT,
                            const __bf16* __restrict__ kB,
                            const __bf16* __restrict__ vB,
                            __bf16* __restrict__ combined) {
  const int h    = blockIdx.y;
  const int mb   = blockIdx.x;
  const int wave = threadIdx.x >> 5;
  const int lane = threadIdx.x & 31;
  const int m0   = mb * 128 + wave * 16;
  __shared__ float scaleS[8][16];

  // preload q^T B-fragments (lane<->m column, 16 features per lane in table order)
  BF16x16 bq[4];
#pragma unroll
  for (int s = 0; s < 4; s++) {
    const __bf16* p = qT + ((size_t)h * HEAD_D + s * 32 + (lane >> 4) * 8) * S_LEN + m0 + (lane & 15);
#pragma unroll
    for (int j = 0; j < 8; j++) {
      bq[s].v[j]     = p[(size_t)j * S_LEN];
      bq[s].v[8 + j] = p[(size_t)(16 + j) * S_LEN];
    }
  }

  f32x8 out[8];
#pragma unroll
  for (int f = 0; f < 8; f++) out[f] = zero8();
  float run_max = -1e30f, run_sum = 0.f;
  const int kRow  = lane & 15;
  const int fHalf = (lane >> 4) * 8;

  for (int kb = 0; kb < S_LEN / 32; kb++) {
    const int key0 = kb * 32;
    f32x8 st0 = zero8(), st1 = zero8();
#pragma unroll
    for (int s = 0; s < 4; s++) {
      BF16x16 ak0, ak1;
      const __bf16* p0 = kB + ((size_t)h * S_LEN + key0 + kRow) * HEAD_D
                            + s * 32 + fHalf;
      ak0.h[0] = *(const bf16x8*)p0;
      ak0.h[1] = *(const bf16x8*)(p0 + 16);
      const __bf16* p1 = p0 + 16 * HEAD_D;
      ak1.h[0] = *(const bf16x8*)p1;
      ak1.h[1] = *(const bf16x8*)(p1 + 16);
      st0 = wmma_bf16(ak0.v, bq[s].v, st0);
      st1 = wmma_bf16(ak1.v, bq[s].v, st1);
    }
    // online softmax over the key (VGPR+lane-pair) dimension
    float bmax = -1e30f;
#pragma unroll
    for (int r = 0; r < 8; r++) {
      bmax = fmaxf(bmax, st0[r]);
      bmax = fmaxf(bmax, st1[r]);
    }
    bmax = fmaxf(bmax, __shfl_xor(bmax, 16, 32));
    const float nmax  = fmaxf(run_max, bmax);
    const float alpha = __expf(run_max - nmax);
    float bsum = 0.f;
    BF16x16 pa;
#pragma unroll
    for (int r = 0; r < 8; r++) {
      float e0 = __expf(st0[r] - nmax);
      float e1 = __expf(st1[r] - nmax);
      bsum += e0 + e1;
      pa.v[r]     = (__bf16)e0;   // tile0 -> A elems 0..7
      pa.v[8 + r] = (__bf16)e1;   // tile1 -> A elems 8..15
    }
    bsum += __shfl_xor(bsum, 16, 32);
    run_sum = run_sum * alpha + bsum;
    run_max = nmax;

    if (lane < 16) scaleS[wave][lane] = alpha;   // alpha per query row m
    float al[8];
#pragma unroll
    for (int r = 0; r < 8; r++) al[r] = scaleS[wave][r + 8 * (lane >> 4)];

#pragma unroll
    for (int f = 0; f < 8; f++) {
      BF16x16 bv;
      const __bf16* vp = vB + ((size_t)h * S_LEN + key0 + (lane >> 4) * 8) * HEAD_D + f * 16 + (lane & 15);
#pragma unroll
      for (int j = 0; j < 8; j++) {
        bv.v[j]     = vp[j * HEAD_D];
        bv.v[8 + j] = vp[(16 + j) * HEAD_D];
      }
      f32x8 o = out[f];
#pragma unroll
      for (int r = 0; r < 8; r++) o[r] *= al[r];
      out[f] = wmma_bf16(pa.v, bv.v, o);
    }
  }

  if (lane < 16) scaleS[wave][lane] = 1.f / run_sum;
  float rs[8];
#pragma unroll
  for (int r = 0; r < 8; r++) rs[r] = scaleS[wave][r + 8 * (lane >> 4)];
#pragma unroll
  for (int f = 0; f < 8; f++)
#pragma unroll
    for (int r = 0; r < 8; r++) {
      const int m = m0 + r + 8 * (lane >> 4);
      combined[(size_t)m * COMB_D + h * HEAD_D + f * 16 + (lane & 15)] =
          (__bf16)(out[f][r] * rs[r]);
    }
}

// =====================================================================
// K6: out = x + gate * (combined @ w2.T + b2)
// =====================================================================
__global__ void gemm2_kernel(const __bf16* __restrict__ combined,
                             const float* __restrict__ w2,
                             const float* __restrict__ b2,
                             const float* __restrict__ x,
                             const float* __restrict__ mod,
                             float* __restrict__ out) {
  __shared__ __bf16 ldsB[2][BK][LDSB_P];
  f32x8 acc[2][4];
  gemm_core(combined, w2, COMB_D, blockIdx.y, blockIdx.x, ldsB, acc);

  const int lane = threadIdx.x & 31;
  const int wave = threadIdx.x >> 5;
  const int mBase = blockIdx.y * BM + (wave & 3) * 32 + 8 * (lane >> 4);
  const int nBase = blockIdx.x * BN + (wave >> 2) * 64 + (lane & 15);
#pragma unroll
  for (int mi = 0; mi < 2; mi++)
#pragma unroll
    for (int ni = 0; ni < 4; ni++) {
      const int n = nBase + ni * 16;
      const float bias = b2[n];
      const float gate = mod[2 * HIDDEN + n];
#pragma unroll
      for (int r = 0; r < 8; r++) {
        const int m = mBase + mi * 16 + r;
        const size_t idx = (size_t)m * HIDDEN + n;
        out[idx] = x[idx] + gate * (acc[mi][ni][r] + bias);
      }
    }
}

// =====================================================================
// launch
// =====================================================================
extern "C" void kernel_launch(void* const* d_in, const int* in_sizes, int n_in,
                              void* d_out, int out_size, void* d_ws, size_t ws_size,
                              hipStream_t stream) {
  const float* x     = (const float*)d_in[0];
  const float* vec   = (const float*)d_in[1];
  const float* cosT  = (const float*)d_in[2];
  const float* sinT  = (const float*)d_in[3];
  const float* w_mod = (const float*)d_in[4];
  const float* b_mod = (const float*)d_in[5];
  const float* w1    = (const float*)d_in[6];
  const float* b1    = (const float*)d_in[7];
  const float* w2    = (const float*)d_in[8];
  const float* b2    = (const float*)d_in[9];
  const float* qw    = (const float*)d_in[10];
  const float* kw    = (const float*)d_in[11];
  float* out = (float*)d_out;

  char* ws = (char*)d_ws;
  size_t off = 0;
  auto alloc = [&](size_t bytes) -> void* {
    void* p = ws + off;
    off += (bytes + 255) & ~(size_t)255;
    return p;
  };
  float*  mod      = (float*) alloc((size_t)QKV_D * 4);
  __bf16* xmod     = (__bf16*)alloc((size_t)S_LEN * HIDDEN * 2);
  float*  qkv      = (float*) alloc((size_t)S_LEN * QKV_D * 4);
  __bf16* qT       = (__bf16*)alloc((size_t)N_HEADS * HEAD_D * S_LEN * 2);
  __bf16* kB       = (__bf16*)alloc((size_t)N_HEADS * S_LEN * HEAD_D * 2);
  __bf16* vB       = (__bf16*)alloc((size_t)N_HEADS * S_LEN * HEAD_D * 2);
  __bf16* combined = (__bf16*)alloc((size_t)S_LEN * COMB_D * 2);
  (void)ws_size; (void)in_sizes; (void)n_in; (void)out_size;

  mod_gemv_kernel<<<dim3(QKV_D / 8), 256, 0, stream>>>(vec, w_mod, b_mod, mod);
  ln_mod_kernel<<<dim3(S_LEN), 256, 0, stream>>>(x, mod, xmod);
  gemm1_kernel<<<dim3(H1_D / BN, S_LEN / BM), 256, 0, stream>>>(xmod, w1, b1, qkv, combined);
  qk_rope_kernel<<<dim3(S_LEN, N_HEADS), 128, 0, stream>>>(qkv, qw, kw, cosT, sinT, qT, kB, vB);
  attn_kernel<<<dim3(S_LEN / 128, N_HEADS), 256, 0, stream>>>(qT, kB, vB, combined);
  gemm2_kernel<<<dim3(HIDDEN / BN, S_LEN / BM), 256, 0, stream>>>(combined, w2, b2, x, mod, out);
}
